// CrossAttentionN_446676599074
// MI455X (gfx1250) — matrix operand order repaired and checked by reading, not compiled, not run on hardware
//
#include <hip/hip_runtime.h>

typedef __bf16 bf16;
typedef __attribute__((ext_vector_type(16))) __bf16 bf16x16;
typedef __attribute__((ext_vector_type(8)))  float  f32x8;

#define B_  32
#define T_  64
#define NJ  22
#define D_  512
#define H_  8
#define C_  128
#define DH  64

// ---------------------------------------------------------------------------
// WMMA helpers (CDNA5 gfx1250, wave32).
// D = A(16x32 bf16) * B(32x16 bf16) + C(16x16 f32)
// A layout: lane l -> row M=l%16; element e -> K = (e/8)*16 + (l/16)*8 + e%8
// B layout: lane l -> col N=l%16; element e -> K = (e/8)*16 + (l/16)*8 + e%8
// C layout: vgpr r -> M = r + 8*(l/16); N = l%16
// ---------------------------------------------------------------------------
__device__ __forceinline__ f32x8 wmma_bf16(bf16x16 a, bf16x16 b, f32x8 c) {
  return __builtin_amdgcn_wmma_f32_16x16x32_bf16(false, a, false, b, (short)0, c,
                                                 false, false);
}

// Load A fragment (16x32) from LDS tile with element pitch `pitch` (pitch%8==0
// so every 8-element group is 16B aligned).
__device__ __forceinline__ bf16x16 load_a_frag(const bf16* base, int pitch,
                                               int row0, int k0) {
  int lane = threadIdx.x & 31;
  int m = lane & 15, half = lane >> 4;
  const bf16* p = base + (row0 + m) * pitch + k0 + half * 8;
  union { uint4 u[2]; bf16x16 v; } un;
  un.u[0] = *reinterpret_cast<const uint4*>(p);        // K = half*8 + 0..7
  un.u[1] = *reinterpret_cast<const uint4*>(p + 16);   // K = 16 + half*8 + 0..7
  return un.v;
}

// Load B fragment (32x16): lane l reads col N=col0+l%16, rows K=k0+(l/16)*8+0..7 and k0+16+(l/16)*8+0..7.
__device__ __forceinline__ bf16x16 load_b_frag(const bf16* base, int pitch,
                                               int k0, int col0) {
  int lane = threadIdx.x & 31;
  int n = lane & 15, half = lane >> 4;
  const bf16* p = base + (k0 + half * 8) * pitch + col0 + n;
  bf16x16 v;
#pragma unroll
  for (int e = 0; e < 8; ++e) {
    v[e] = p[e * pitch];
    v[8 + e] = p[(16 + e) * pitch];
  }
  return v;
}

// ---------------------------------------------------------------------------
// CDNA5 async global->LDS copy (16B per active lane), tracked by ASYNCcnt.
// Inline asm (bypasses toolchain-dependent builtin arity).  LDS address is the
// low 32 bits of the generic pointer (LDS aperture keeps offset in addr[31:0]).
// GV addressing: 64-bit global address in a VGPR pair, saddr = off.
// ---------------------------------------------------------------------------
__device__ __forceinline__ void async_copy_b128(void* lds, const void* g) {
  unsigned       l  = (unsigned)(unsigned long long)(size_t)lds;
  unsigned long long ga = (unsigned long long)(size_t)g;
  asm volatile("global_load_async_to_lds_b128 %0, %1, off"
               :: "v"(l), "v"(ga) : "memory");
}
__device__ __forceinline__ void wait_async0() {
  asm volatile("s_wait_asynccnt 0x0" ::: "memory");
}

// ---------------------------------------------------------------------------
// f32 -> bf16 conversion (all sizes are multiples of 4)
// ---------------------------------------------------------------------------
__global__ void __launch_bounds__(256)
cvt_f32_to_bf16(const float* __restrict__ in, bf16* __restrict__ out, long n) {
  long i = ((long)blockIdx.x * blockDim.x + threadIdx.x) * 4;
  long stride = (long)gridDim.x * blockDim.x * 4;
  for (; i < n; i += stride) {
    float4 f = *reinterpret_cast<const float4*>(in + i);
    out[i + 0] = (bf16)f.x;
    out[i + 1] = (bf16)f.y;
    out[i + 2] = (bf16)f.z;
    out[i + 3] = (bf16)f.w;
  }
}

// ---------------------------------------------------------------------------
// Tiled bf16 GEMM, C[M,512] = A[M,512] * W[512,512] + bias.  N=K=512 fixed.
// Block tile 128x128, BK=32, 256 threads = 8 waves arranged 4(M) x 2(N);
// each wave owns a 32x64 output patch (2x4 WMMA tiles).
// Double-buffered LDS fed by async global->LDS DMA: issue tile kk+1, compute
// tile kk, pay s_wait_asynccnt 0 + barrier at the boundary.
// blockIdx.z batches (per-joint Wq).  OUT_F32: 0 -> bf16 out, 1 -> f32 out.
// M must be a multiple of 128 (true for all uses: 2048 / 4096 / 45056).
// ---------------------------------------------------------------------------
template <int OUT_F32>
__global__ void __launch_bounds__(256)
gemm512_bf16(const bf16* __restrict__ Aall, long lda, long abatch,
             const bf16* __restrict__ Wall, long wbatch,
             const float* __restrict__ biasall, long bbatch,
             void* __restrict__ Call, long ldc, long cbatch) {
  __shared__ bf16 As[2][128 * 40];   // pitch 40 (=BK+8): 16B-aligned frags
  __shared__ bf16 Bs[2][32 * 136];   // pitch 136 (=BN+8)

  const int z = blockIdx.z;
  const bf16* A = Aall + (size_t)z * abatch;
  const bf16* W = Wall + (size_t)z * wbatch;
  const float* bias = biasall + (size_t)z * bbatch;
  const int row0 = blockIdx.x * 128;
  const int col0 = blockIdx.y * 128;
  const int tid = threadIdx.x;
  const int w = tid >> 5, lane = tid & 31;
  const int wm = w & 3, wn = w >> 2;
  const int half = lane >> 4, nl = lane & 15;

  // Async-stage one BK=32 tile pair (A: 128x32, W: 32x128), 4 x b128 / thread.
  auto stage = [&](int buf, int k0) {
#pragma unroll
    for (int i = 0; i < 2; ++i) {
      int idx = tid + i * 256;
      int r = idx >> 2, j = idx & 3;
      async_copy_b128(&As[buf][r * 40 + j * 8],
                      A + (size_t)(row0 + r) * lda + k0 + j * 8);
    }
#pragma unroll
    for (int i = 0; i < 2; ++i) {
      int idx = tid + i * 256;
      int r = idx >> 4, j = idx & 15;
      async_copy_b128(&Bs[buf][r * 136 + j * 8],
                      W + (size_t)(k0 + r) * 512 + col0 + j * 8);
    }
  };

  f32x8 acc[2][4];
#pragma unroll
  for (int mt = 0; mt < 2; ++mt)
#pragma unroll
    for (int nt = 0; nt < 4; ++nt) acc[mt][nt] = (f32x8)0.0f;

  stage(0, 0);
  for (int kk = 0; kk < 16; ++kk) {
    wait_async0();       // this wave's async copies (incl. next-buf) done
    __syncthreads();     // -> everyone's copies into buf kk are visible
    if (kk + 1 < 16) stage((kk + 1) & 1, (kk + 1) * 32);
    const int buf = kk & 1;
    bf16x16 a0 = load_a_frag(As[buf], 40, wm * 32, 0);
    bf16x16 a1 = load_a_frag(As[buf], 40, wm * 32 + 16, 0);
#pragma unroll
    for (int nt = 0; nt < 4; ++nt) {
      bf16x16 bfr = load_b_frag(Bs[buf], 136, 0, wn * 64 + nt * 16);
      acc[0][nt] = wmma_bf16(a0, bfr, acc[0][nt]);
      acc[1][nt] = wmma_bf16(a1, bfr, acc[1][nt]);
    }
    __syncthreads();     // all reads of buf kk done before it is re-written
  }

#pragma unroll
  for (int mt = 0; mt < 2; ++mt) {
    int grow0 = row0 + wm * 32 + mt * 16 + half * 8;
#pragma unroll
    for (int nt = 0; nt < 4; ++nt) {
      int gcol = col0 + wn * 64 + nt * 16 + nl;
      float bv = bias[gcol];
      if (OUT_F32) {
        float* Cf = reinterpret_cast<float*>(Call) + (size_t)z * cbatch;
#pragma unroll
        for (int r = 0; r < 8; ++r)
          Cf[(size_t)(grow0 + r) * ldc + gcol] = acc[mt][nt][r] + bv;
      } else {
        bf16* Cb = reinterpret_cast<bf16*>(Call) + (size_t)z * cbatch;
#pragma unroll
        for (int r = 0; r < 8; ++r)
          Cb[(size_t)(grow0 + r) * ldc + gcol] = (bf16)(acc[mt][nt][r] + bv);
      }
    }
  }
}

// ---------------------------------------------------------------------------
// Fused attention: one block per (b, n, h).  4 waves, wave w owns query rows
// 16w..16w+15 so the softmax over C=128 is wave-local (16-lane reductions).
//   S = Qh (64x64) * Kh^T (64x128), P = softmax(S/8), O = P * Vh (128x64)
// Q/V tiles arrive via async global->LDS DMA; K is transposed via VGPRs.
// ---------------------------------------------------------------------------
__global__ void __launch_bounds__(128)
attn_fused(const bf16* __restrict__ qbf, const bf16* __restrict__ kbf,
           const bf16* __restrict__ vbf, bf16* __restrict__ obf) {
  __shared__ bf16 qs[64 * 72];    // [t][dh]   pitch 72
  __shared__ bf16 ks[64 * 136];   // K^T: [dh][c] pitch 136
  __shared__ bf16 vs[128 * 72];   // [c][dh]   pitch 72
  __shared__ bf16 ps[64 * 136];   // P: [t][c] pitch 136
  const int h = blockIdx.x, n = blockIdx.y, b = blockIdx.z;
  const int tid = threadIdx.x;
  const int w = tid >> 5, lane = tid & 31, half = lane >> 4, nl = lane & 15;

  // Q tile (64x64): 512 async b128, 4/thread.
  const bf16* qg = qbf + (((size_t)b * T_) * NJ + n) * D_ + h * DH;
#pragma unroll
  for (int i = 0; i < 4; ++i) {
    int idx = tid + i * 128;
    int r = idx >> 3, j = idx & 7;
    async_copy_b128(&qs[r * 72 + j * 8], qg + (size_t)r * (NJ * D_) + j * 8);
  }
  // V tile (128x64): 1024 async b128, 8/thread.
  const bf16* vg = vbf + (size_t)b * C_ * D_ + h * DH;
#pragma unroll
  for (int i = 0; i < 8; ++i) {
    int idx = tid + i * 128;
    int r = idx >> 3, j = idx & 7;
    async_copy_b128(&vs[r * 72 + j * 8], vg + (size_t)r * D_ + j * 8);
  }
  // K tile, transposed into ks[dh][c] (element-wise via VGPRs).
  const bf16* kg = kbf + (size_t)b * C_ * D_ + h * DH;
  for (int idx = tid; idx < C_ * DH; idx += 128) {
    int c = idx >> 6, d = idx & 63;
    ks[d * 136 + c] = kg[(size_t)c * D_ + d];
  }
  wait_async0();
  __syncthreads();

  // S = Q * K^T : 8 column tiles, K-dim = 64 (2 WMMA steps).
  f32x8 s[8];
#pragma unroll
  for (int ct = 0; ct < 8; ++ct) s[ct] = (f32x8)0.0f;
#pragma unroll
  for (int kk = 0; kk < 2; ++kk) {
    bf16x16 a = load_a_frag(qs, 72, 16 * w, kk * 32);
#pragma unroll
    for (int ct = 0; ct < 8; ++ct)
      s[ct] = wmma_bf16(a, load_b_frag(ks, 136, kk * 32, ct * 16), s[ct]);
  }

  // Row softmax (rows are wave-local; reduce over 8 tiles + 16 lanes).
  const float scale = 0.125f;  // 1/sqrt(64)
  float mx[8], rs[8];
#pragma unroll
  for (int r = 0; r < 8; ++r) mx[r] = -1e30f;
#pragma unroll
  for (int ct = 0; ct < 8; ++ct)
#pragma unroll
    for (int r = 0; r < 8; ++r) {
      s[ct][r] *= scale;
      mx[r] = fmaxf(mx[r], s[ct][r]);
    }
#pragma unroll
  for (int r = 0; r < 8; ++r) {
#pragma unroll
    for (int off = 1; off < 16; off <<= 1)
      mx[r] = fmaxf(mx[r], __shfl_xor(mx[r], off, 16));
    rs[r] = 0.0f;
  }
#pragma unroll
  for (int ct = 0; ct < 8; ++ct)
#pragma unroll
    for (int r = 0; r < 8; ++r) {
      float p = __expf(s[ct][r] - mx[r]);
      rs[r] += p;
      ps[(16 * w + r + 8 * half) * 136 + ct * 16 + nl] = (bf16)p;
    }
#pragma unroll
  for (int r = 0; r < 8; ++r)
#pragma unroll
    for (int off = 1; off < 16; off <<= 1)
      rs[r] += __shfl_xor(rs[r], off, 16);
  __syncthreads();

  // O = P * V : 4 output tiles over dh, K-dim = 128 (4 WMMA steps).
  f32x8 o[4];
#pragma unroll
  for (int nt = 0; nt < 4; ++nt) o[nt] = (f32x8)0.0f;
#pragma unroll
  for (int kk = 0; kk < 4; ++kk) {
    bf16x16 a = load_a_frag(ps, 136, 16 * w, kk * 32);
#pragma unroll
    for (int nt = 0; nt < 4; ++nt)
      o[nt] = wmma_bf16(a, load_b_frag(vs, 72, kk * 32, nt * 16), o[nt]);
  }
  float inv[8];
#pragma unroll
  for (int r = 0; r < 8; ++r) inv[r] = 1.0f / rs[r];
#pragma unroll
  for (int nt = 0; nt < 4; ++nt)
#pragma unroll
    for (int r = 0; r < 8; ++r) {
      int t = 16 * w + r + 8 * half;
      obf[(((size_t)b * T_ + t) * NJ + n) * D_ + h * DH + nt * 16 + nl] =
          (bf16)(o[nt][r] * inv[r]);
    }
}

// ---------------------------------------------------------------------------
// Launcher: convert -> Qproj (batched over joints) -> K,V -> attention -> out.
// Workspace use: ~157 MiB of bf16 staging in d_ws.
// ---------------------------------------------------------------------------
extern "C" void kernel_launch(void* const* d_in, const int* in_sizes, int n_in,
                              void* d_out, int out_size, void* d_ws, size_t ws_size,
                              hipStream_t stream) {
  (void)in_sizes; (void)n_in; (void)out_size; (void)ws_size;
  const float* x       = (const float*)d_in[0];
  const float* context = (const float*)d_in[1];
  const float* Wq      = (const float*)d_in[2];
  const float* bq      = (const float*)d_in[3];
  const float* Wk      = (const float*)d_in[4];
  const float* bk      = (const float*)d_in[5];
  const float* Wv      = (const float*)d_in[6];
  const float* bv      = (const float*)d_in[7];
  const float* Wout    = (const float*)d_in[8];
  const float* bout    = (const float*)d_in[9];
  float* out = (float*)d_out;

  char* ws = (char*)d_ws;
  size_t off = 0;
  auto alloc = [&](size_t elems) {
    bf16* p = (bf16*)(ws + off);
    off += ((elems * sizeof(bf16) + 255) / 256) * 256;
    return p;
  };
  const size_t nx   = (size_t)B_ * T_ * NJ * D_;  // 23.1M
  const size_t nctx = (size_t)B_ * C_ * D_;       // 2.1M
  bf16* xbf   = alloc(nx);
  bf16* qbf   = alloc(nx);
  bf16* obf   = alloc(nx);
  bf16* ctxbf = alloc(nctx);
  bf16* kbf   = alloc(nctx);
  bf16* vbf   = alloc(nctx);
  bf16* wqbf  = alloc((size_t)NJ * D_ * D_);
  bf16* wkbf  = alloc((size_t)D_ * D_);
  bf16* wvbf  = alloc((size_t)D_ * D_);
  bf16* wobf  = alloc((size_t)D_ * D_);

  auto cvt = [&](const float* src, bf16* dst, long n) {
    int blocks = (int)((n / 4 + 255) / 256);
    if (blocks > 4096) blocks = 4096;
    cvt_f32_to_bf16<<<blocks, 256, 0, stream>>>(src, dst, n);
  };
  cvt(x, xbf, (long)nx);
  cvt(context, ctxbf, (long)nctx);
  cvt(Wq, wqbf, (long)NJ * D_ * D_);
  cvt(Wk, wkbf, (long)D_ * D_);
  cvt(Wv, wvbf, (long)D_ * D_);
  cvt(Wout, wobf, (long)D_ * D_);

  // Stage A: q[b,t,n,:] = x[b,t,n,:] @ Wq[n] + bq[n]   (batched over joints)
  gemm512_bf16<0><<<dim3((B_ * T_) / 128, 4, NJ), 256, 0, stream>>>(
      xbf, (long)NJ * D_, (long)D_, wqbf, (long)D_ * D_, bq, (long)D_,
      qbf, (long)NJ * D_, (long)D_);
  // Stage B: k = context @ Wk + bk ; v = context @ Wv + bv
  gemm512_bf16<0><<<dim3((B_ * C_) / 128, 4, 1), 256, 0, stream>>>(
      ctxbf, D_, 0, wkbf, 0, bk, 0, kbf, D_, 0);
  gemm512_bf16<0><<<dim3((B_ * C_) / 128, 4, 1), 256, 0, stream>>>(
      ctxbf, D_, 0, wvbf, 0, bv, 0, vbf, D_, 0);
  // Stage C: attention per (b, n, h)
  attn_fused<<<dim3(H_, NJ, B_), 128, 0, stream>>>(qbf, kbf, vbf, obf);
  // Stage D: out = o @ Wout + bout  (f32 output)
  gemm512_bf16<1><<<dim3((B_ * T_ * NJ) / 128, 4, 1), 256, 0, stream>>>(
      obf, D_, 0, wobf, 0, bout, 0, (void*)out, D_, 0);
}
